// CSnT_51290499449095
// MI455X (gfx1250) — matrix operand order repaired and checked by reading, not compiled, not run on hardware
//
#include <hip/hip_runtime.h>

typedef __attribute__((ext_vector_type(4)))  float   v4f;
typedef __attribute__((ext_vector_type(8)))  float   v8f;
typedef __attribute__((ext_vector_type(8)))  __bf16  v8bf;
typedef __attribute__((ext_vector_type(16))) __bf16  v16bf;

#define M_TOT 16384   // B*S = 8*2048
#define N_TOT 512     // H
#define K_TOT 512     // IN
#define BM 128
#define BN 64
#define BK 32
#define KSTEPS (K_TOT / BK)          // 16
#define LDSS 40                      // padded bf16 row stride (80B, 16B aligned)
#define OUT_PLANE ((size_t)M_TOT * N_TOT)  // 8388608

__global__ __launch_bounds__(256)
void snn_fused_wmma(const float* __restrict__ x,
                    const float* __restrict__ W,
                    const float* __restrict__ gNaP,
                    const float* __restrict__ gKP,
                    const float* __restrict__ gLP,
                    float* __restrict__ out)
{
    __shared__ __bf16 ldsA[2][BM * LDSS];   // 2 * 10240 B
    __shared__ __bf16 ldsB[2][BN * LDSS];   // 2 *  5120 B

    const int tid  = threadIdx.x;
    const int lane = tid & 31;
    const int wave = tid >> 5;
    const int n0   = blockIdx.x * BN;
    const int m0   = blockIdx.y * BM;

    const int waveM = (wave & 3) * 32;   // 4 waves along M
    const int waveN = (wave >> 2) * 32;  // 2 waves along N

    // ---- global->LDS staging assignments (all 256 threads) ----
    const int aRow = tid >> 1;               // 0..127
    const int aCol = (tid & 1) * 16;         // 0 or 16
    const float* gA = x + (size_t)(m0 + aRow) * K_TOT + aCol;

    const int bRow = tid >> 2;               // 0..63
    const int bCol = (tid & 3) * 8;          // 0,8,16,24
    const float* gB = W + (size_t)(n0 + bRow) * K_TOT + bCol;

    auto stage = [&](int buf, int kt) {
        const float* pa = gA + kt * BK;
        v4f a0 = *(const v4f*)(pa + 0);
        v4f a1 = *(const v4f*)(pa + 4);
        v4f a2 = *(const v4f*)(pa + 8);
        v4f a3 = *(const v4f*)(pa + 12);
        v8bf lo, hi;
#pragma unroll
        for (int i = 0; i < 4; ++i) {
            lo[i]     = (__bf16)a0[i];
            lo[i + 4] = (__bf16)a1[i];
            hi[i]     = (__bf16)a2[i];
            hi[i + 4] = (__bf16)a3[i];
        }
        __bf16* da = &ldsA[buf][aRow * LDSS + aCol];
        *(v8bf*)(da + 0) = lo;
        *(v8bf*)(da + 8) = hi;

        const float* pb = gB + kt * BK;
        v4f b0 = *(const v4f*)(pb + 0);
        v4f b1 = *(const v4f*)(pb + 4);
        v8bf q;
#pragma unroll
        for (int i = 0; i < 4; ++i) {
            q[i]     = (__bf16)b0[i];
            q[i + 4] = (__bf16)b1[i];
        }
        *(v8bf*)(&ldsB[buf][bRow * LDSS + bCol]) = q;
    };

    v8f c00 = {}, c01 = {}, c10 = {}, c11 = {};

    stage(0, 0);
    __syncthreads();

    const int mrow = lane & 15;
    const int aK0  = (lane < 16) ? 0 : 8;      // A lane K-chunk base
    const int bK0  = (lane >> 4) * 8;          // B lane K base

#pragma unroll 1
    for (int kt = 0; kt < KSTEPS; ++kt) {
        const int cur = kt & 1;
        if (kt + 1 < KSTEPS) stage(cur ^ 1, kt + 1);

        // A fragments: lane holds row m, K = {aK0..aK0+7} U {aK0+16..aK0+23}
        v16bf afr[2];
#pragma unroll
        for (int ti = 0; ti < 2; ++ti) {
            const __bf16* base = &ldsA[cur][(waveM + ti * 16 + mrow) * LDSS];
            v8bf lo = *(const v8bf*)(base + aK0);
            v8bf hi = *(const v8bf*)(base + aK0 + 16);
            afr[ti] = __builtin_shufflevector(lo, hi,
                0,1,2,3,4,5,6,7,8,9,10,11,12,13,14,15);
        }
        // B fragments: lane holds col n, K = {bK0..bK0+7} U {bK0+16..bK0+23}
        v16bf bfr[2];
#pragma unroll
        for (int tj = 0; tj < 2; ++tj) {
            const __bf16* base = &ldsB[cur][(waveN + tj * 16 + mrow) * LDSS + bK0];
            v8bf lo = *(const v8bf*)(base + 0);
            v8bf hi = *(const v8bf*)(base + 16);
            bfr[tj] = __builtin_shufflevector(lo, hi,
                0,1,2,3,4,5,6,7,8,9,10,11,12,13,14,15);
        }

        c00 = __builtin_amdgcn_wmma_f32_16x16x32_bf16(false, afr[0], false, bfr[0], (short)0, c00, false, false);
        c01 = __builtin_amdgcn_wmma_f32_16x16x32_bf16(false, afr[0], false, bfr[1], (short)0, c01, false, false);
        c10 = __builtin_amdgcn_wmma_f32_16x16x32_bf16(false, afr[1], false, bfr[0], (short)0, c10, false, false);
        c11 = __builtin_amdgcn_wmma_f32_16x16x32_bf16(false, afr[1], false, bfr[1], (short)0, c11, false, false);

        __syncthreads();
    }

    // ---- fused HH / adaptive-LIF epilogue on accumulator registers ----
    const float gNa = gNaP[0];
    const float gK  = gKP[0];
    const float gL  = gLP[0];

    auto emit = [&](const v8f& c, int ti, int tj) {
        const int mBase = m0 + waveM + ti * 16 + ((lane >> 4) * 8);
        const int nIdx  = n0 + waveN + tj * 16 + (lane & 15);
#pragma unroll
        for (int r = 0; r < 8; ++r) {
            float Vv = c[r];                                   // psp == V
            float am = 0.1f * (Vv + 40.f) / (1.f - __expf(-(Vv + 40.f) * 0.1f));
            float bm = 4.f * __expf(-(Vv + 65.f) * (1.f / 18.f));
            float ah = 0.07f * __expf(-(Vv + 65.f) * 0.05f);
            float bh = 1.f / (1.f + __expf(-(Vv + 35.f) * 0.1f));
            float an = 0.01f * (Vv + 55.f) / (1.f - __expf(-(Vv + 55.f) * 0.1f));
            float bn = 0.125f * __expf(-(Vv + 65.f) * (1.f / 80.f));
            float mg = 0.05f + 0.1f * (am * (1.f - 0.05f) - bm * 0.05f);
            float hg = 0.60f + 0.1f * (ah * (1.f - 0.60f) - bh * 0.60f);
            float ng = 0.32f + 0.1f * (an * (1.f - 0.32f) - bn * 0.32f);
            float Iion = gNa * mg * mg * mg * hg * (Vv - 50.f)
                       + gK * ng * ng * ng * ng * (Vv + 77.f)
                       + gL * (Vv + 54.4f);
            float Iin   = Iion + Vv;
            float vnew  = -65.f + Iin * (0.1f / 20.f);
            float spike = (vnew >= -50.f) ? 1.f : 0.f;
            float wnew  = (0.5f * (vnew + 65.f) + 0.1f * spike) * (0.1f / 100.f);
            float vrs   = (spike > 0.5f) ? -65.f : vnew;

            const size_t idx = (size_t)(mBase + r) * N_TOT + nIdx;
            out[idx]                 = spike;
            out[idx + OUT_PLANE]     = vrs;
            out[idx + 2 * OUT_PLANE] = wnew;
        }
    };
    emit(c00, 0, 0); emit(c01, 0, 1); emit(c10, 1, 0); emit(c11, 1, 1);
}

extern "C" void kernel_launch(void* const* d_in, const int* in_sizes, int n_in,
                              void* d_out, int out_size, void* d_ws, size_t ws_size,
                              hipStream_t stream) {
    const float* x   = (const float*)d_in[0];
    const float* W   = (const float*)d_in[1];
    const float* gNa = (const float*)d_in[2];
    const float* gK  = (const float*)d_in[3];
    const float* gL  = (const float*)d_in[4];
    float* out = (float*)d_out;

    dim3 grid(N_TOT / BN, M_TOT / BM);   // (8, 128) = 1024 workgroups
    snn_fused_wmma<<<grid, 256, 0, stream>>>(x, W, gNa, gK, gL, out);
    (void)in_sizes; (void)n_in; (void)out_size; (void)d_ws; (void)ws_size;
}
